// SegmentedAttention_14379550507469
// MI455X (gfx1250) — matrix operand order repaired and checked by reading, not compiled, not run on hardware
//
#include <hip/hip_runtime.h>
#include <hip/hip_bf16.h>

// ---------------------------------------------------------------------------
// MI455X (gfx1250, wave32) segmented attention.
// Matmuls via v_wmma_f32_16x16x32_bf16 (fp32 accumulate); GEMM B-tiles staged
// in LDS through the CDNA5 async global->LDS engine (ASYNCcnt).
// ---------------------------------------------------------------------------

typedef __bf16 bf16_t;
typedef __attribute__((ext_vector_type(16))) __bf16 bf16x16;
typedef __attribute__((ext_vector_type(8)))  __bf16 bf16x8;
typedef __attribute__((ext_vector_type(8)))  float  f32x8;

#define BATCH  2
#define NSEQ   4096
#define DIM    1024
#define HEADS  16
#define DH     64
#define SEG    512
#define WSEG   8                 // NSEQ / SEG
#define NSEGS  (BATCH * WSEG)    // 16 folded segments
#define PM     4
#define KPAD   544               // 17 * 32-key chunks (PM+SEG=516 padded)
#define QKSCALE 0.125f           // DH^-0.5
#define NEGBIG (-1e30f)

static __device__ __forceinline__ f32x8 zero8() {
  f32x8 z;
#pragma unroll
  for (int i = 0; i < 8; ++i) z[i] = 0.f;
  return z;
}

static __device__ __forceinline__ bf16x16 make_frag(bf16x8 lo, bf16x8 hi) {
  bf16x16 r;
#pragma unroll
  for (int i = 0; i < 8; ++i) { r[i] = lo[i]; r[i + 8] = hi[i]; }
  return r;
}

// A-matrix fragment (16x32 bf16): lane = row (l&15); elements are two
// contiguous 8-elem runs at K = 8*(l>>4) and 16 + 8*(l>>4).
static __device__ __forceinline__ bf16x16 load_a_frag(const bf16_t* rowptr, int h16) {
  bf16x8 lo = *(const bf16x8*)(rowptr + 8 * h16);
  bf16x8 hi = *(const bf16x8*)(rowptr + 16 + 8 * h16);
  return make_frag(lo, hi);
}

// B-matrix fragment (32x16 bf16): lane = N column (l&15); elements are K rows
// 8*(l>>4)+s and 16+8*(l>>4)+s (stride ld), matching the A fragment.
static __device__ __forceinline__ bf16x16 load_b_frag(const bf16_t* p, int ld, int h16) {
  bf16x16 r;
#pragma unroll
  for (int i = 0; i < 8; ++i) {
    r[i] = p[(size_t)(8 * h16 + i) * ld];
    r[i + 8] = p[(size_t)(16 + 8 * h16 + i) * ld];
  }
  return r;
}

static __device__ __forceinline__ f32x8 wmma_bf16(bf16x16 a, bf16x16 b, f32x8 c) {
  return __builtin_amdgcn_wmma_f32_16x16x32_bf16(false, a, false, b, (short)0, c,
                                                 false, false);
}

// CDNA5 async global->LDS copy, 16B per lane, tracked by ASYNCcnt.
// lds_off: per-lane LDS byte offset (low 32 bits of generic shared pointer).
static __device__ __forceinline__ void async_ld_b128(unsigned int lds_off,
                                                     const bf16_t* gptr) {
  asm volatile("global_load_async_to_lds_b128 %0, %1, off"
               :
               : "v"(lds_off), "v"(gptr)
               : "memory");
}
static __device__ __forceinline__ void wait_async0() {
  asm volatile("s_wait_asynccnt 0" ::: "memory");
}
static __device__ __forceinline__ void wait_ds0() {
  asm volatile("s_wait_dscnt 0" ::: "memory");
}

// ---------------------------------------------------------------------------
// RMSNorm: one row (1024 elems) per 256-thread block, bf16 output.
// ---------------------------------------------------------------------------
__global__ __launch_bounds__(256) void k_rmsnorm(const float* __restrict__ seq,
                                                 const float* __restrict__ g,
                                                 bf16_t* __restrict__ xn) {
  const int row = blockIdx.x;
  const float4 v = ((const float4*)(seq + (size_t)row * DIM))[threadIdx.x];
  __shared__ float red[256];
  red[threadIdx.x] = v.x * v.x + v.y * v.y + v.z * v.z + v.w * v.w;
  __syncthreads();
  for (int s = 128; s > 0; s >>= 1) {
    if (threadIdx.x < s) red[threadIdx.x] += red[threadIdx.x + s];
    __syncthreads();
  }
  const float scale = rsqrtf(red[0] * (1.f / DIM) + 1e-6f);
  const float4 gg = ((const float4*)g)[threadIdx.x];
  bf16_t* o = xn + (size_t)row * DIM + threadIdx.x * 4;
  o[0] = (bf16_t)(v.x * scale * gg.x);
  o[1] = (bf16_t)(v.y * scale * gg.y);
  o[2] = (bf16_t)(v.z * scale * gg.z);
  o[3] = (bf16_t)(v.w * scale * gg.w);
}

__global__ __launch_bounds__(256) void k_cvt(const float* __restrict__ in,
                                             bf16_t* __restrict__ out, int n) {
  const int i = blockIdx.x * 256 + threadIdx.x;
  if (i < n) out[i] = (bf16_t)in[i];
}

// ---------------------------------------------------------------------------
// WMMA GEMM: C[M x NN] = A[M x 1024](bf16) * Bw[1024 x NN](bf16).
// Block tile 128M x 128N (8 waves = 4M x 2N; each wave 32x64 = 2x4 wmma tiles).
// Per K-step (32): the 32x128 B tile (8KB) is async-staged to LDS once per
// block (global_load_async_to_lds_b128 + s_wait_asynccnt), then all waves read
// fragments with ds loads.  MODE 0: fp32 row-major out.  MODE 1: QKV scatter.
// ---------------------------------------------------------------------------
template <int NN, int MODE>
__global__ __launch_bounds__(256) void k_gemm(const bf16_t* __restrict__ A,
                                              const bf16_t* __restrict__ Bw,
                                              float* __restrict__ outf,
                                              bf16_t* __restrict__ qb,
                                              bf16_t* __restrict__ kb,
                                              bf16_t* __restrict__ vb,
                                              float* __restrict__ vout) {
  __shared__ bf16_t ldsB[32][128];   // [k_local][n_local], 8KB
  const int tid = threadIdx.x;
  const int lane = tid & 31;
  const int l16 = lane & 15, h16 = lane >> 4;
  const int warp = tid >> 5;
  const int wm_l = warp & 3, wn_l = warp >> 2;
  const int nblk = NN / 128;
  const int m0 = (blockIdx.x / nblk) * 128 + wm_l * 32;
  const int n0b = (blockIdx.x % nblk) * 128;
  const int n0 = n0b + wn_l * 64;

  // Async staging assignment: 512 x 16B chunks per K-step, 2 per thread.
  const int c0 = tid * 2, c1 = tid * 2 + 1;
  const int r0 = c0 >> 4, col0 = (c0 & 15) * 8;   // 8 bf16 = 16B
  const int r1 = c1 >> 4, col1 = (c1 & 15) * 8;
  const unsigned int loff0 = (unsigned int)(size_t)&ldsB[r0][col0];
  const unsigned int loff1 = (unsigned int)(size_t)&ldsB[r1][col1];
  const bf16_t* g0 = Bw + (size_t)r0 * NN + n0b + col0;
  const bf16_t* g1 = Bw + (size_t)r1 * NN + n0b + col1;

  f32x8 acc[2][4];
#pragma unroll
  for (int mt = 0; mt < 2; ++mt)
#pragma unroll
    for (int nt = 0; nt < 4; ++nt) acc[mt][nt] = zero8();

  const bf16_t* arow0 = A + (size_t)(m0 + l16) * DIM;
  const bf16_t* arow1 = A + (size_t)(m0 + 16 + l16) * DIM;
  const bf16_t* lbase = &ldsB[0][wn_l * 64 + l16];

  for (int kk = 0; kk < DIM; kk += 32) {
    async_ld_b128(loff0, g0);
    async_ld_b128(loff1, g1);
    g0 += (size_t)32 * NN;
    g1 += (size_t)32 * NN;
    const bf16x16 a0 = load_a_frag(arow0 + kk, h16);
    const bf16x16 a1 = load_a_frag(arow1 + kk, h16);
    if (kk + 32 < DIM) {  // warm next A rows (global_prefetch_b8)
      __builtin_prefetch(arow0 + kk + 32, 0, 0);
      __builtin_prefetch(arow1 + kk + 32, 0, 0);
    }
    wait_async0();        // this wave's async deposits done
    __syncthreads();      // whole 32x128 tile visible in LDS
#pragma unroll
    for (int nt = 0; nt < 4; ++nt) {
      const bf16x16 b = load_b_frag(lbase + nt * 16, 128, h16);   // ds loads, lane = N
      acc[0][nt] = wmma_bf16(a0, b, acc[0][nt]);
      acc[1][nt] = wmma_bf16(a1, b, acc[1][nt]);
    }
    wait_ds0();
    __syncthreads();      // protect LDS tile from next iteration's deposits
  }

  // C layout: lane l, VGPR r -> (M = r + 8*(l>>4), N = l&15)
#pragma unroll
  for (int mt = 0; mt < 2; ++mt)
#pragma unroll
    for (int nt = 0; nt < 4; ++nt)
#pragma unroll
      for (int r = 0; r < 8; ++r) {
        const int m = m0 + mt * 16 + r + 8 * h16;
        const int c = n0 + nt * 16 + l16;
        const float v = acc[mt][nt][r];
        if (MODE == 0) {
          outf[(size_t)m * NN + c] = v;
        } else {
          const int which = c >> 10;              // 0=q 1=k 2=v
          const int cc = c & 1023;
          const int hh = cc >> 6, dd = cc & 63;
          const int bb = m >> 12, nn = m & 4095;
          const size_t idx =
              (((size_t)(bb * HEADS + hh)) * NSEQ + nn) * DH + dd;
          const bf16_t vb16 = (bf16_t)v;
          if (which == 0) qb[idx] = vb16;
          else if (which == 1) kb[idx] = vb16;
          else { vb[idx] = vb16; vout[idx] = v; }   // orig_v fp32 output
        }
      }
}

// ---------------------------------------------------------------------------
// Interleaved RoPE in-place on q,k (bf16), one (b,h,n,pair) per thread.
// ---------------------------------------------------------------------------
__global__ __launch_bounds__(256) void k_rope(bf16_t* __restrict__ q,
                                              bf16_t* __restrict__ k) {
  const int idx = blockIdx.x * 256 + threadIdx.x;   // B*H*N*(DH/2) threads
  const int p = idx & 31;
  const int n = (idx >> 5) & (NSEQ - 1);
  const int bh = idx >> 17;                          // b*HEADS + h
  const float inv = __powf(10000.f, -(float)(2 * p) * (1.f / DH));
  float s, c;
  __sincosf((float)n * inv, &s, &c);
  const size_t base = ((size_t)bh * NSEQ + n) * DH + 2 * p;
  const float q0 = (float)q[base], q1 = (float)q[base + 1];
  q[base]     = (bf16_t)(q0 * c - q1 * s);
  q[base + 1] = (bf16_t)(q1 * c + q0 * s);
  const float k0 = (float)k[base], k1 = (float)k[base + 1];
  k[base]     = (bf16_t)(k0 * c - k1 * s);
  k[base + 1] = (bf16_t)(k1 * c + k0 * s);
}

// ---------------------------------------------------------------------------
// Build padded per-segment KV: ktpad[s,h,dh,KPAD] (dh-major for QK^T B-frags)
// and vpad[s,h,KPAD,dh] (key-major for PV B-frags). j<PM from pmem, tail = 0.
// ---------------------------------------------------------------------------
__global__ __launch_bounds__(256) void k_pads(const bf16_t* __restrict__ kb,
                                              const bf16_t* __restrict__ vb,
                                              const float* __restrict__ pmem,
                                              bf16_t* __restrict__ ktpad,
                                              bf16_t* __restrict__ vpad) {
  const int idx = blockIdx.x * 256 + threadIdx.x;  // NSEGS*HEADS*KPAD*DH
  const int dd = idx & 63;
  const int j = (idx >> 6) % KPAD;
  const int sh = idx / (64 * KPAD);
  const int s = sh >> 4, h = sh & 15;
  const int b = s >> 3, w = s & 7;
  float kvk = 0.f, kvv = 0.f;
  if (j < PM) {
    const size_t pidx = (((size_t)h) * PM + j) * DH + dd;
    kvk = pmem[pidx];
    kvv = pmem[(size_t)HEADS * PM * DH + pidx];
  } else if (j < PM + SEG) {
    const int n = w * SEG + (j - PM);
    const size_t gi = (((size_t)(b * HEADS + h)) * NSEQ + n) * DH + dd;
    kvk = (float)kb[gi];
    kvv = (float)vb[gi];
  }
  ktpad[((size_t)sh * DH + dd) * KPAD + j] = (bf16_t)kvk;
  vpad[((size_t)sh * KPAD + j) * DH + dd]  = (bf16_t)kvv;
}

// ---------------------------------------------------------------------------
// Flash-style attention: one wave owns a 16-row q tile of one (segment,head).
// Per 32-key chunk: 4 score WMMAs (K=dh=64 split in two), online softmax via
// shfl_xor row reductions, P re-layout through per-wave LDS, 4 PV WMMAs.
// ---------------------------------------------------------------------------
__global__ __launch_bounds__(256) void k_attn(const bf16_t* __restrict__ q,
                                              const bf16_t* __restrict__ ktpad,
                                              const bf16_t* __restrict__ vpad,
                                              bf16_t* __restrict__ obuf) {
  __shared__ bf16_t plds[8][16][32];
  const int lane = threadIdx.x & 31, l16 = lane & 15, h16 = lane >> 4;
  const int warp = threadIdx.x >> 5;
  const int wid = blockIdx.x * 8 + warp;
  const int qt = wid & 31;                 // 32 q-tiles per (seg,head)
  const int h = (wid >> 5) & 15;
  const int s = wid >> 9;                  // folded segment 0..15
  const int b = s >> 3, w = s & 7;
  const int qi0 = qt * 16;

  const bf16_t* qrow =
      q + (((size_t)(b * HEADS + h)) * NSEQ + (w * SEG + qi0 + l16)) * DH;
  const bf16x16 qa0 = load_a_frag(qrow, h16);        // dh 0..31
  const bf16x16 qa1 = load_a_frag(qrow + 32, h16);   // dh 32..63

  const bf16_t* ktb = ktpad + ((size_t)(s * HEADS + h)) * DH * KPAD;
  const bf16_t* vpb = vpad + ((size_t)(s * HEADS + h)) * KPAD * DH;

  f32x8 oacc[4];
#pragma unroll
  for (int t = 0; t < 4; ++t) oacc[t] = zero8();
  float m_r[8], l_r[8];
#pragma unroll
  for (int r = 0; r < 8; ++r) { m_r[r] = NEGBIG; l_r[r] = 0.f; }

  const int nch = (PM + qi0 + 16 + 31) >> 5;   // causal chunk bound (uniform)
  for (int cidx = 0; cidx < nch; ++cidx) {
    const int kb0 = cidx * 32;
    // QK^T: B column = key (l16), elements = dh rows (stride KPAD) from dh-major Kt.
    const bf16_t* kt0 = ktb + kb0 + l16;
    const bf16_t* kt1 = ktb + (size_t)32 * KPAD + kb0 + l16;
    f32x8 s0 = zero8(), s1 = zero8();
    s0 = wmma_bf16(qa0, load_b_frag(kt0, KPAD, h16), s0);
    s0 = wmma_bf16(qa1, load_b_frag(kt1, KPAD, h16), s0);
    s1 = wmma_bf16(qa0, load_b_frag(kt0 + 16, KPAD, h16), s1);
    s1 = wmma_bf16(qa1, load_b_frag(kt1 + 16, KPAD, h16), s1);

    const int jk0 = kb0 + l16, jk1 = kb0 + 16 + l16;
#pragma unroll
    for (int r = 0; r < 8; ++r) {
      const int irow = qi0 + r + 8 * h16;
      float v0 = s0[r] * QKSCALE;
      float v1 = s1[r] * QKSCALE;
      v0 = ((jk0 < PM) || (jk0 - PM <= irow)) ? v0 : NEGBIG;
      v1 = ((jk1 < PM) || (jk1 - PM <= irow)) ? v1 : NEGBIG;
      float mx = fmaxf(v0, v1);
      mx = fmaxf(mx, __shfl_xor(mx, 1, 32));
      mx = fmaxf(mx, __shfl_xor(mx, 2, 32));
      mx = fmaxf(mx, __shfl_xor(mx, 4, 32));
      mx = fmaxf(mx, __shfl_xor(mx, 8, 32));
      const float mnew = fmaxf(m_r[r], mx);
      const float alpha = __expf(m_r[r] - mnew);
      const float p0 = __expf(v0 - mnew);
      const float p1 = __expf(v1 - mnew);
      float rs = p0 + p1;
      rs += __shfl_xor(rs, 1, 32);
      rs += __shfl_xor(rs, 2, 32);
      rs += __shfl_xor(rs, 4, 32);
      rs += __shfl_xor(rs, 8, 32);
      l_r[r] = l_r[r] * alpha + rs;
      m_r[r] = mnew;
#pragma unroll
      for (int t = 0; t < 4; ++t) oacc[t][r] *= alpha;
      plds[warp][r + 8 * h16][l16]      = (bf16_t)p0;   // C-layout -> LDS
      plds[warp][r + 8 * h16][16 + l16] = (bf16_t)p1;
    }
    // In-wave LDS RAW: DS ops are in-order per wave; fence the compiler + DS.
    wait_ds0();
    const bf16x8 plo = *(const bf16x8*)&plds[warp][l16][8 * h16];
    const bf16x8 phi = *(const bf16x8*)&plds[warp][l16][16 + 8 * h16];
    const bf16x16 pf = make_frag(plo, phi);             // A-layout P frag
#pragma unroll
    for (int t = 0; t < 4; ++t) {
      const bf16x16 vf = load_b_frag(vpb + (size_t)kb0 * DH + 16 * t + l16, DH, h16);
      oacc[t] = wmma_bf16(pf, vf, oacc[t]);
    }
    wait_ds0();
  }

  const size_t orow0 = (size_t)b * NSEQ + w * SEG + qi0;
#pragma unroll
  for (int r = 0; r < 8; ++r) {
    const float inv = 1.f / l_r[r];
    const size_t grow = (orow0 + r + 8 * h16) * DIM + h * DH;
#pragma unroll
    for (int t = 0; t < 4; ++t)
      obuf[grow + 16 * t + l16] = (bf16_t)(oacc[t][r] * inv);
  }
}

// ---------------------------------------------------------------------------
extern "C" void kernel_launch(void* const* d_in, const int* in_sizes, int n_in,
                              void* d_out, int out_size, void* d_ws,
                              size_t ws_size, hipStream_t stream) {
  (void)in_sizes; (void)n_in; (void)out_size; (void)ws_size;
  const float* seq   = (const float*)d_in[0];
  const float* normg = (const float*)d_in[1];
  const float* wqkv  = (const float*)d_in[2];
  const float* wout  = (const float*)d_in[3];
  const float* pmem  = (const float*)d_in[4];
  float* out  = (float*)d_out;
  float* vout = out + (size_t)BATCH * NSEQ * DIM;   // orig_v, [b,h,n,dh] fp32

  char* ws = (char*)d_ws;
  auto take = [&](size_t bytes) {
    char* p = ws;
    ws += (bytes + 255) & ~(size_t)255;
    return p;
  };
  bf16_t* xn     = (bf16_t*)take((size_t)BATCH * NSEQ * DIM * 2);
  bf16_t* wqkv_b = (bf16_t*)take((size_t)DIM * 3 * DIM * 2);
  bf16_t* wout_b = (bf16_t*)take((size_t)DIM * DIM * 2);
  bf16_t* qb     = (bf16_t*)take((size_t)BATCH * HEADS * NSEQ * DH * 2);
  bf16_t* kb     = (bf16_t*)take((size_t)BATCH * HEADS * NSEQ * DH * 2);
  bf16_t* vb     = (bf16_t*)take((size_t)BATCH * HEADS * NSEQ * DH * 2);
  bf16_t* ktpad  = (bf16_t*)take((size_t)NSEGS * HEADS * DH * KPAD * 2);
  bf16_t* vpad   = (bf16_t*)take((size_t)NSEGS * HEADS * KPAD * DH * 2);
  bf16_t* obuf   = (bf16_t*)take((size_t)BATCH * NSEQ * DIM * 2);

  k_rmsnorm<<<BATCH * NSEQ, 256, 0, stream>>>(seq, normg, xn);
  k_cvt<<<(DIM * 3 * DIM) / 256, 256, 0, stream>>>(wqkv, wqkv_b, DIM * 3 * DIM);
  k_cvt<<<(DIM * DIM) / 256, 256, 0, stream>>>(wout, wout_b, DIM * DIM);

  // QKV GEMM: 8192x3072 in 128x128 block tiles -> 64*24 = 1536 blocks.
  k_gemm<3 * DIM, 1><<<(BATCH * NSEQ / 128) * (3 * DIM / 128), 256, 0,
                       stream>>>(xn, wqkv_b, nullptr, qb, kb, vb, vout);

  k_rope<<<(BATCH * HEADS * NSEQ * (DH / 2)) / 256, 256, 0, stream>>>(qb, kb);
  k_pads<<<((size_t)NSEGS * HEADS * KPAD * DH) / 256, 256, 0, stream>>>(
      kb, vb, pmem, ktpad, vpad);

  // Attention: 16 segs * 16 heads * 32 q-tiles = 8192 waves -> 1024 blocks.
  k_attn<<<(NSEGS * HEADS * (SEG / 16)) / 8, 256, 0, stream>>>(qb, ktpad, vpad,
                                                               obuf);

  // Output GEMM: 8192x1024 in 128x128 block tiles -> 64*8 = 512 blocks.
  k_gemm<DIM, 0><<<(BATCH * NSEQ / 128) * (DIM / 128), 256, 0, stream>>>(
      obuf, wout_b, out, nullptr, nullptr, nullptr, nullptr);
}
